// ByteWaveletTransformer_82729660056319
// MI455X (gfx1250) — matrix operand order repaired and checked by reading, not compiled, not run on hardware
//
#include <hip/hip_runtime.h>
#include <hip/hip_bf16.h>
#include <math.h>

// ---------------------------------------------------------------------------
// ByteWaveletTransformer forward for gfx1250 (MI455X).
// All GEMMs run through v_wmma_f32_16x16x32_bf16 (bf16 in, f32 accumulate),
// software-pipelined with double-buffered fragments.
// ---------------------------------------------------------------------------

typedef __bf16 bf16;
typedef __bf16 v16bf __attribute__((ext_vector_type(16)));
typedef float  v8f   __attribute__((ext_vector_type(8)));

static constexpr int   DM   = 512;    // model dim
static constexpr int   NH   = 8;      // heads
static constexpr int   HD   = 64;     // head dim
static constexpr int   NL   = 4;      // layers
static constexpr int   FF   = 2048;   // ffn dim
static constexpr int   NS   = 11;     // wavelet scales
static constexpr int   SEQ  = 2048;
static constexpr int   BATCH= 4;
static constexpr int   TOK  = BATCH * SEQ;       // 8192
static constexpr int   NDEC = 16 * 257;          // 4112

__device__ __constant__ float D4C[4] = {
  0.4829629131445341f, 0.8365163037378079f,
  0.2241438680420134f, -0.1294095225512604f };

static __device__ __forceinline__ bf16 f2bf(float f) {
  union { float f; unsigned u; } a; a.f = f;
  unsigned r = (a.u + 0x7FFFu + ((a.u >> 16) & 1u)) >> 16;
  union { unsigned short s; bf16 b; } o; o.s = (unsigned short)r;
  return o.b;
}
static __device__ __forceinline__ float gelu_exact(float v) {
  return 0.5f * v * (1.0f + erff(v * 0.7071067811865476f));
}
static __device__ __forceinline__ float sigmoidf(float v) {
  return 1.0f / (1.0f + __expf(-v));
}

// 16-lane-striped bf16 fragment (8 VGPRs); two 16B chunks OFF2 elements apart
union Frag { uint4 u[2]; v16bf v; };
template <int OFF2>
static __device__ __forceinline__ void load_frag(Frag& f, const bf16* p) {
  f.u[0] = *reinterpret_cast<const uint4*>(p);
  f.u[1] = *reinterpret_cast<const uint4*>(p + OFF2);
}

// ---------------------------------------------------------------------------
// f32 -> bf16 conversion (weights)
// ---------------------------------------------------------------------------
__global__ __launch_bounds__(256) void f2bf_kernel(const float* __restrict__ src,
                                                   bf16* __restrict__ dst, int count) {
  int i = blockIdx.x * 256 + threadIdx.x;
  if (i < count) dst[i] = f2bf(src[i]);
}

// ---------------------------------------------------------------------------
// softmax prep: gains = softmax(scale_gain, axis=S), cps = softmax(coupling, -1)
// ---------------------------------------------------------------------------
__global__ void prep_kernel(const float* __restrict__ sg, const float* __restrict__ cp,
                            float* __restrict__ gains, float* __restrict__ cps) {
  int t = threadIdx.x;
  if (t >= NL * NH) return;
  int l = t >> 3, h = t & 7;
  float mx = -1e30f;
  for (int j = 0; j < NS; ++j) mx = fmaxf(mx, sg[l * NS * NH + j * NH + h]);
  float e[NS], s = 0.f;
  for (int j = 0; j < NS; ++j) { e[j] = __expf(sg[l * NS * NH + j * NH + h] - mx); s += e[j]; }
  for (int j = 0; j < NS; ++j) gains[l * NS * NH + j * NH + h] = e[j] / s;
  float mx2 = -1e30f;
  for (int j = 0; j < NH; ++j) mx2 = fmaxf(mx2, cp[l * 64 + h * 8 + j]);
  float e2[NH], s2 = 0.f;
  for (int j = 0; j < NH; ++j) { e2[j] = __expf(cp[l * 64 + h * 8 + j] - mx2); s2 += e2[j]; }
  for (int j = 0; j < NH; ++j) cps[l * 64 + h * 8 + j] = e2[j] / s2;
}

// ---------------------------------------------------------------------------
// Byte patch encoder: one block per token.
// ---------------------------------------------------------------------------
__global__ __launch_bounds__(256) void encoder_kernel(
    const int* __restrict__ patches, const float* __restrict__ byte_emb,
    const float* __restrict__ conv_w, const float* __restrict__ conv_b,
    const float* __restrict__ enc_w, const float* __restrict__ enc_b,
    const float* __restrict__ pos_emb, float* __restrict__ x) {
  __shared__ float emb[16][32];
  __shared__ float hcv[16][32];
  __shared__ float mean[32];
  __shared__ float cw[3 * 32 * 32];
  const int tok = blockIdx.x;
  const int n = tok & (SEQ - 1);
  const int* pp = patches + (size_t)tok * 16;
  for (int i = threadIdx.x; i < 3 * 32 * 32; i += 256) cw[i] = conv_w[i];
  for (int e = threadIdx.x; e < 512; e += 256) {
    int p = e >> 5, c = e & 31;
    int id = pp[p];
    emb[p][c] = (id > 0) ? byte_emb[id * 32 + c] : 0.f;
  }
  __syncthreads();
  for (int e = threadIdx.x; e < 512; e += 256) {
    int p = e >> 5, co = e & 31;
    float s = conv_b[co];
    for (int dp = 0; dp < 3; ++dp) {
      int ps = p + dp - 1;
      if (ps >= 0 && ps < 16)
        for (int ci = 0; ci < 32; ++ci) s += emb[ps][ci] * cw[dp * 1024 + ci * 32 + co];
    }
    hcv[p][co] = gelu_exact(s);
  }
  __syncthreads();
  if (threadIdx.x < 32) {
    int c = threadIdx.x;
    float s = 0.f, mct = 0.f;
    for (int p = 0; p < 16; ++p) {
      float m = (pp[p] > 0) ? 1.f : 0.f;
      s += hcv[p][c] * m;
      mct += m;
    }
    mean[c] = s / (mct + 1e-8f);
  }
  __syncthreads();
  for (int e = threadIdx.x; e < DM; e += 256) {
    float s = enc_b[e];
    for (int ci = 0; ci < 32; ++ci) s += mean[ci] * enc_w[e * 32 + ci];
    x[(size_t)tok * DM + e] = s + pos_emb[(size_t)n * DM + e];
  }
}

// ---------------------------------------------------------------------------
// LayerNorm over 512 features; writes bf16 (always) and f32 (if outF != null)
// ---------------------------------------------------------------------------
__global__ __launch_bounds__(256) void ln_kernel(
    const float* __restrict__ x, const float* __restrict__ g,
    const float* __restrict__ bta, bf16* __restrict__ outB,
    float* __restrict__ outF) {
  __shared__ float red[8];
  __shared__ float stats[2];
  const int row = blockIdx.x;
  const float* xr = x + (size_t)row * DM;
  const int c0 = threadIdx.x, c1 = threadIdx.x + 256;
  float v0 = xr[c0], v1 = xr[c1];
  float s = v0 + v1;
  for (int o = 16; o > 0; o >>= 1) s += __shfl_down(s, o, 32);
  int lane = threadIdx.x & 31, wv = threadIdx.x >> 5;
  if (lane == 0) red[wv] = s;
  __syncthreads();
  if (threadIdx.x == 0) {
    float t = 0.f;
    for (int i = 0; i < 8; ++i) t += red[i];
    stats[0] = t * (1.f / DM);
  }
  __syncthreads();
  float mean = stats[0];
  float d0 = v0 - mean, d1 = v1 - mean;
  float q = d0 * d0 + d1 * d1;
  for (int o = 16; o > 0; o >>= 1) q += __shfl_down(q, o, 32);
  if (lane == 0) red[wv] = q;
  __syncthreads();
  if (threadIdx.x == 0) {
    float t = 0.f;
    for (int i = 0; i < 8; ++i) t += red[i];
    stats[1] = rsqrtf(t * (1.f / DM) + 1e-5f);
  }
  __syncthreads();
  float rstd = stats[1];
  float o0 = d0 * rstd * g[c0] + bta[c0];
  float o1 = d1 * rstd * g[c1] + bta[c1];
  outB[(size_t)row * DM + c0] = f2bf(o0);
  outB[(size_t)row * DM + c1] = f2bf(o1);
  if (outF) {
    outF[(size_t)row * DM + c0] = o0;
    outF[(size_t)row * DM + c1] = o1;
  }
}

// ---------------------------------------------------------------------------
// WMMA GEMM:  C[M,N] = A[M,K](bf16) * W[N,K](bf16)^T + bias, fused epilogue.
// Block = 256 threads = 8 waves; each wave owns a 16 x (NSUB*16) strip of C.
// Double-buffered fragments (distinct VGPRs) so the ~10 loads of the next
// K-step overlap the 4 WMMAs of the current one (partial s_wait_loadcnt
// instead of full drains). K must be a multiple of 64.
// EPI: 0=store f32, 1=gelu->bf16, 2=sigmoid->f32, 3=resid+=, 4=resid+=mul*
// ---------------------------------------------------------------------------
template <int EPI, int NSUB>
__global__ __launch_bounds__(256) void wmma_gemm(
    const bf16* __restrict__ A, const bf16* __restrict__ W,
    const float* __restrict__ bias, float* __restrict__ outF,
    bf16* __restrict__ outB, const float* __restrict__ mul,
    float* __restrict__ resid, int M, int N, int K, int colbase) {
  const int lane = threadIdx.x & 31;
  const int wave = threadIdx.x >> 5;
  const int lr = lane & 15;
  const int hi = lane >> 4;
  const int m0 = blockIdx.y * 128 + wave * 16;
  const int nbase = colbase + blockIdx.x * (NSUB * 16);

  v8f acc[NSUB];
#pragma unroll
  for (int s = 0; s < NSUB; ++s)
#pragma unroll
    for (int r = 0; r < 8; ++r) acc[s][r] = 0.f;

  const bf16* ap = A + (size_t)(m0 + lr) * K + hi * 16;
  const bf16* wp[NSUB];
#pragma unroll
  for (int s = 0; s < NSUB; ++s)
    wp[s] = W + (size_t)(nbase + s * 16 + lr) * K + hi * 16;

  Frag a0, a1, b0[NSUB], b1[NSUB];
  // prologue: stage 0 at k=0
  load_frag<8>(a0, ap);
#pragma unroll
  for (int s = 0; s < NSUB; ++s) load_frag<8>(b0[s], wp[s]);

  int k = 0;
  for (; k + 64 < K; k += 64) {
    // issue stage-1 loads (k+32), then consume stage 0
    load_frag<8>(a1, ap + k + 32);
#pragma unroll
    for (int s = 0; s < NSUB; ++s) load_frag<8>(b1[s], wp[s] + k + 32);
    __builtin_prefetch((const void*)(ap + k + 128), 0, 1);
#pragma unroll
    for (int s = 0; s < NSUB; ++s)
      acc[s] = __builtin_amdgcn_wmma_f32_16x16x32_bf16(
          false, a0.v, false, b0[s].v, (short)0, acc[s], false, false);
    // issue stage-0 loads (k+64), then consume stage 1
    load_frag<8>(a0, ap + k + 64);
#pragma unroll
    for (int s = 0; s < NSUB; ++s) load_frag<8>(b0[s], wp[s] + k + 64);
#pragma unroll
    for (int s = 0; s < NSUB; ++s)
      acc[s] = __builtin_amdgcn_wmma_f32_16x16x32_bf16(
          false, a1.v, false, b1[s].v, (short)0, acc[s], false, false);
  }
  // tail: last 64 of K (K % 64 == 0)
  load_frag<8>(a1, ap + k + 32);
#pragma unroll
  for (int s = 0; s < NSUB; ++s) load_frag<8>(b1[s], wp[s] + k + 32);
#pragma unroll
  for (int s = 0; s < NSUB; ++s)
    acc[s] = __builtin_amdgcn_wmma_f32_16x16x32_bf16(
        false, a0.v, false, b0[s].v, (short)0, acc[s], false, false);
#pragma unroll
  for (int s = 0; s < NSUB; ++s)
    acc[s] = __builtin_amdgcn_wmma_f32_16x16x32_bf16(
        false, a1.v, false, b1[s].v, (short)0, acc[s], false, false);

#pragma unroll
  for (int s = 0; s < NSUB; ++s) {
    const int col = nbase + s * 16 + lr;
    const float bv = bias ? bias[col] : 0.f;
#pragma unroll
    for (int r = 0; r < 8; ++r) {
      const int row = m0 + hi * 8 + r;
      const size_t idx = (size_t)row * N + col;
      float vv = acc[s][r] + bv;
      if (EPI == 0) outF[idx] = vv;
      else if (EPI == 1) outB[idx] = f2bf(gelu_exact(vv));
      else if (EPI == 2) outF[idx] = sigmoidf(vv);
      else if (EPI == 3) resid[idx] += vv;
      else if (EPI == 4) resid[idx] += mul[idx] * vv;
    }
  }
}

// ---------------------------------------------------------------------------
// ||k|| per (token, head): one wave per head vector of 64.
// kv layout: [TOK][1024], k = cols 0..511, v = cols 512..1023.
// ---------------------------------------------------------------------------
__global__ __launch_bounds__(256) void knorm_kernel(const float* __restrict__ kv,
                                                    float* __restrict__ kn) {
  int w = blockIdx.x * 8 + (threadIdx.x >> 5);
  int lane = threadIdx.x & 31;
  int tok = w >> 3, h = w & 7;
  const float* kp = kv + (size_t)tok * 1024 + h * HD;
  float a = kp[lane], b = kp[lane + 32];
  float s = a * a + b * b;
  for (int o = 16; o > 0; o >>= 1) s += __shfl_down(s, o, 32);
  if (lane == 0) kn[(size_t)tok * NH + h] = sqrtf(s);
}

__global__ __launch_bounds__(256) void field_kernel(const float* __restrict__ kv,
                                                    const float* __restrict__ kn,
                                                    float* __restrict__ field) {
  int i = blockIdx.x * 256 + threadIdx.x;
  int tok = i >> 9, c = i & 511;
  field[i] = kv[(size_t)tok * 1024 + 512 + c] * kn[tok * NH + (c >> 6)];
}

// ---------------------------------------------------------------------------
// Causal dilated D4 wavelet across 11 scales with softmaxed gains.
// ---------------------------------------------------------------------------
__global__ __launch_bounds__(256) void wavelet_kernel(const float* __restrict__ field,
                                                      const float* __restrict__ gains,
                                                      float* __restrict__ accb) {
  int i = blockIdx.x * 256 + threadIdx.x;
  int t = i >> 9, c = i & 511;
  int n = t & (SEQ - 1);
  int h = c >> 6;
  const float fcur = field[i];
  float a = 0.f;
#pragma unroll
  for (int j = 0; j < NS; ++j) {
    int d = 1 << j;
    float y = D4C[3] * fcur;
    if (n >= d)     y += D4C[2] * field[(size_t)(t - d) * DM + c];
    if (n >= 2 * d) y += D4C[1] * field[(size_t)(t - 2 * d) * DM + c];
    if (n >= 3 * d) y += D4C[0] * field[(size_t)(t - 3 * d) * DM + c];
    a += gains[j * NH + h] * y;
  }
  accb[i] = a;
}

// head coupling (8x8) then multiply with sigmoid gate -> bf16 GEMM input
__global__ __launch_bounds__(256) void combine_kernel(const float* __restrict__ accb,
                                                      const float* __restrict__ cp,
                                                      const float* __restrict__ gate,
                                                      bf16* __restrict__ gg) {
  int i = blockIdx.x * 256 + threadIdx.x;
  int c = i & 511, h = c >> 6, hd = c & 63;
  const float* ap = accb + (size_t)(i >> 9) * DM + hd;
  float f = 0.f;
#pragma unroll
  for (int j = 0; j < NH; ++j) f += cp[h * 8 + j] * ap[j * HD];
  gg[i] = f2bf(f * gate[i]);
}

// causal mean pool via cumsum; one thread per (batch, channel)
__global__ __launch_bounds__(256) void cumsum_kernel(const float* __restrict__ xi,
                                                     bf16* __restrict__ pool) {
  int t = blockIdx.x * 256 + threadIdx.x;   // 0..2047
  int b = t >> 9, c = t & 511;
  float s = 0.f;
  for (int n = 0; n < SEQ; ++n) {
    size_t idx = (size_t)(b * SEQ + n) * DM + c;
    s += xi[idx];
    pool[idx] = f2bf(s / (float)(n + 1));
  }
}

// ---------------------------------------------------------------------------
// Host-side launch
// ---------------------------------------------------------------------------
extern "C" void kernel_launch(void* const* d_in, const int* in_sizes, int n_in,
                              void* d_out, int out_size, void* d_ws, size_t ws_size,
                              hipStream_t stream) {
  const float* byte_emb = (const float*)d_in[0];
  const float* conv_w   = (const float*)d_in[1];
  const float* conv_b   = (const float*)d_in[2];
  const float* enc_w    = (const float*)d_in[3];
  const float* enc_b    = (const float*)d_in[4];
  const float* pos_emb  = (const float*)d_in[5];
  const float* ln1_g    = (const float*)d_in[6];
  const float* ln1_b    = (const float*)d_in[7];
  const float* qkv_w    = (const float*)d_in[8];
  const float* qkv_b    = (const float*)d_in[9];
  const float* out_w    = (const float*)d_in[10];
  const float* out_b    = (const float*)d_in[11];
  const float* gate_w   = (const float*)d_in[12];
  const float* gate_b   = (const float*)d_in[13];
  const float* scale_g  = (const float*)d_in[14];
  const float* coupl    = (const float*)d_in[15];
  const float* ln2_g    = (const float*)d_in[16];
  const float* ln2_b    = (const float*)d_in[17];
  const float* fc1_w    = (const float*)d_in[18];
  const float* fc1_b    = (const float*)d_in[19];
  const float* fc2_w    = (const float*)d_in[20];
  const float* fc2_b    = (const float*)d_in[21];
  const float* int_g    = (const float*)d_in[22];
  const float* int_b    = (const float*)d_in[23];
  const float* ig_w     = (const float*)d_in[24];
  const float* ig_b     = (const float*)d_in[25];
  const float* ip_w     = (const float*)d_in[26];
  const float* ip_b     = (const float*)d_in[27];
  const float* lnf_g    = (const float*)d_in[28];
  const float* lnf_b    = (const float*)d_in[29];
  const float* dec_w    = (const float*)d_in[30];
  const float* dec_b    = (const float*)d_in[31];
  const int*   patches  = (const int*)d_in[32];
  (void)in_sizes; (void)n_in; (void)out_size; (void)ws_size;

  // -------- workspace layout --------
  size_t off = 0;
  auto take = [&](size_t bytes) -> void* {
    void* p = (char*)d_ws + off;
    off += (bytes + 255) & ~(size_t)255;
    return p;
  };
  float* x    = (float*)take((size_t)TOK * DM * 4);
  bf16*  xn   = (bf16*) take((size_t)TOK * DM * 2);
  float* kv   = (float*)take((size_t)TOK * 1024 * 4);
  bf16*  h1   = (bf16*)kv;                       // alias (kv dead before FFN)
  float* kn   = (float*)take((size_t)TOK * NH * 4);
  float* field= (float*)take((size_t)TOK * DM * 4);
  float* xi   = field;                           // alias
  float* accb = (float*)take((size_t)TOK * DM * 4);
  float* sg   = accb;                            // alias
  float* gate = (float*)take((size_t)TOK * DM * 4);
  bf16*  gg   = (bf16*) take((size_t)TOK * DM * 2);
  bf16*  pool = gg;                              // alias
  float* gains= (float*)take(NL * NS * NH * 4);
  float* cps  = (float*)take(NL * NH * NH * 4);
  bf16* qkvW = (bf16*)take((size_t)NL * 3 * DM * DM * 2);
  bf16* outW = (bf16*)take((size_t)NL * DM * DM * 2);
  bf16* gateW= (bf16*)take((size_t)NL * DM * DM * 2);
  bf16* fc1W = (bf16*)take((size_t)NL * FF * DM * 2);
  bf16* fc2W = (bf16*)take((size_t)NL * DM * FF * 2);
  bf16* decW = (bf16*)take((size_t)NDEC * DM * 2);
  bf16* igW  = (bf16*)take((size_t)DM * DM * 2);
  bf16* ipW  = (bf16*)take((size_t)DM * DM * 2);

  auto cvt = [&](const float* s, bf16* d, int cnt) {
    f2bf_kernel<<<(cnt + 255) / 256, 256, 0, stream>>>(s, d, cnt);
  };
  cvt(qkv_w, qkvW, NL * 3 * DM * DM);
  cvt(out_w, outW, NL * DM * DM);
  cvt(gate_w, gateW, NL * DM * DM);
  cvt(fc1_w, fc1W, NL * FF * DM);
  cvt(fc2_w, fc2W, NL * DM * FF);
  cvt(dec_w, decW, NDEC * DM);
  cvt(ig_w, igW, DM * DM);
  cvt(ip_w, ipW, DM * DM);

  prep_kernel<<<1, 64, 0, stream>>>(scale_g, coupl, gains, cps);
  encoder_kernel<<<TOK, 256, 0, stream>>>(patches, byte_emb, conv_w, conv_b,
                                          enc_w, enc_b, pos_emb, x);

  const dim3 blk(256);
  const int gy = TOK / 128;             // 64
  const int ew = TOK * DM / 256;        // elementwise grid

  for (int l = 0; l < NL; ++l) {
    ln_kernel<<<TOK, blk, 0, stream>>>(x, ln1_g + l * DM, ln1_b + l * DM, xn, nullptr);
    // kv = xn @ qkv_w[l, 512:1536].T + qkv_b[l, 512:1536]   (q is unused)
    wmma_gemm<0, 4><<<dim3(1024 / 64, gy), blk, 0, stream>>>(
        xn, qkvW + ((size_t)l * 3 * DM + DM) * DM, qkv_b + l * 3 * DM + DM,
        kv, nullptr, nullptr, nullptr, TOK, 1024, DM, 0);
    knorm_kernel<<<TOK, blk, 0, stream>>>(kv, kn);
    field_kernel<<<ew, blk, 0, stream>>>(kv, kn, field);
    wavelet_kernel<<<ew, blk, 0, stream>>>(field, gains + l * NS * NH, accb);
    // gate = sigmoid(xn @ gate_w[l].T + gate_b[l])
    wmma_gemm<2, 4><<<dim3(DM / 64, gy), blk, 0, stream>>>(
        xn, gateW + (size_t)l * DM * DM, gate_b + l * DM,
        gate, nullptr, nullptr, nullptr, TOK, DM, DM, 0);
    combine_kernel<<<ew, blk, 0, stream>>>(accb, cps + l * 64, gate, gg);
    // x += (coupled * gate) @ out_w[l].T + out_b[l]
    wmma_gemm<3, 4><<<dim3(DM / 64, gy), blk, 0, stream>>>(
        gg, outW + (size_t)l * DM * DM, out_b + l * DM,
        nullptr, nullptr, nullptr, x, TOK, DM, DM, 0);
    if (l == 3) {
      ln_kernel<<<TOK, blk, 0, stream>>>(x, int_g, int_b, xn, xi);
      cumsum_kernel<<<BATCH * DM / 256, blk, 0, stream>>>(xi, pool);
      wmma_gemm<2, 4><<<dim3(DM / 64, gy), blk, 0, stream>>>(
          xn, igW, ig_b, sg, nullptr, nullptr, nullptr, TOK, DM, DM, 0);
      wmma_gemm<4, 4><<<dim3(DM / 64, gy), blk, 0, stream>>>(
          pool, ipW, ip_b, nullptr, nullptr, sg, x, TOK, DM, DM, 0);
    }
    ln_kernel<<<TOK, blk, 0, stream>>>(x, ln2_g + l * DM, ln2_b + l * DM, xn, nullptr);
    // h1 = gelu(xn @ fc1_w[l].T + fc1_b[l]) -> bf16
    wmma_gemm<1, 4><<<dim3(FF / 64, gy), blk, 0, stream>>>(
        xn, fc1W + (size_t)l * FF * DM, fc1_b + l * FF,
        nullptr, h1, nullptr, nullptr, TOK, FF, DM, 0);
    // x += h1 @ fc2_w[l].T + fc2_b[l]
    wmma_gemm<3, 4><<<dim3(DM / 64, gy), blk, 0, stream>>>(
        h1, fc2W + (size_t)l * DM * FF, fc2_b + l * DM,
        nullptr, nullptr, nullptr, x, TOK, DM, FF, 0);
  }
  ln_kernel<<<TOK, blk, 0, stream>>>(x, lnf_g, lnf_b, xn, nullptr);
  // logits = xf @ dec_w.T + dec_b  -> d_out (f32, [B,N,P,VB])
  // main body: columns 0..4095 (64 blocks of 64), tail: columns 4096..4111
  wmma_gemm<0, 4><<<dim3(4096 / 64, gy), blk, 0, stream>>>(
      xn, decW, dec_b, (float*)d_out, nullptr, nullptr, nullptr, TOK, NDEC, DM, 0);
  wmma_gemm<0, 1><<<dim3(1, gy), blk, 0, stream>>>(
      xn, decW, dec_b, (float*)d_out, nullptr, nullptr, nullptr, TOK, NDEC, DM, 4096);
}
